// BiLSTM_CRF_49855980372119
// MI455X (gfx1250) — matrix operand order repaired and checked by reading, not compiled, not run on hardware
//
#include <hip/hip_runtime.h>
#include <hip/hip_bf16.h>

// Problem constants (from reference)
#define V_  50257
#define E_  1024           // embedding dim == 2H (layer 1/2 input dim)
#define H_  512
#define L_  3
#define S_  4096
#define T_  5
#define G_  (4*H_)         // 2048 gate rows per direction
#define NEGINF (-10000.0f)

typedef __attribute__((ext_vector_type(16))) __bf16 v16bf;
typedef __attribute__((ext_vector_type(8)))  float  v8f;

union FragU {
    v16bf v;
    float4 q[2];
};

__device__ __forceinline__ unsigned short f2bf(float f) {
    union { float f; unsigned u; } v; v.f = f;
    unsigned r = v.u + 0x7FFFu + ((v.u >> 16) & 1u);   // round-to-nearest-even
    return (unsigned short)(r >> 16);
}

__device__ __forceinline__ float sigmoidf(float x) {
    return 1.0f / (1.0f + __expf(-x));
}

// CDNA5 async global -> LDS copy (VGLOBAL op 98, tracked by ASYNCcnt).
// Each active lane copies 16 bytes: LDS[ldsoff] = MEM[gaddr].
__device__ __forceinline__ void async_ld_b128(unsigned ldsoff, unsigned long long gaddr) {
    asm volatile("global_load_async_to_lds_b128 %0, %1, off"
                 :: "v"(ldsoff), "v"(gaddr)
                 : "memory");
}
__device__ __forceinline__ void wait_asynccnt0() {
    asm volatile("s_wait_asynccnt 0x0" ::: "memory");
}

// ---------------------------------------------------------------------------
// Embedding gather + f32->bf16 convert:  Xbf[s][e] = bf16(embedding[sentence[s]][e])
// ---------------------------------------------------------------------------
__global__ void k_embed(const int* __restrict__ sent, const float* __restrict__ emb,
                        unsigned short* __restrict__ xbf) {
    int idx = blockIdx.x * blockDim.x + threadIdx.x;
    if (idx >= S_ * E_) return;
    int s = idx >> 10;          // / 1024
    int e = idx & (E_ - 1);
    xbf[idx] = f2bf(emb[(size_t)sent[s] * E_ + e]);
}

// f32 -> bf16 bulk convert (weights, inter-layer activations)
__global__ void k_cvt_bf16(const float* __restrict__ src, unsigned short* __restrict__ dst, int n) {
    int i = blockIdx.x * blockDim.x + threadIdx.x;
    if (i < n) dst[i] = f2bf(src[i]);
}

// Transpose w_hh: [L*2][G][H] -> [L*2][H][G]  (coalesced recurrent reads)
__global__ void k_whh_t(const float* __restrict__ w, float* __restrict__ wt) {
    int i = blockIdx.x * blockDim.x + threadIdx.x;
    const int total = L_ * 2 * G_ * H_;
    if (i >= total) return;
    int k  = i % H_;
    int j  = (i / H_) % G_;
    int ld = i / (H_ * G_);
    wt[(size_t)ld * H_ * G_ + (size_t)k * G_ + j] = w[i];
}

// ---------------------------------------------------------------------------
// WMMA GEMM:  XG[m][n] = sum_k Xb[m][k] * Wb[n][k]  + b_ih[n] + b_hh[n]
//   Xb: [S][E] bf16 (row major), Wb: [G][E] bf16 (row major), XG: [S][G] f32
// Block = 256 threads = 8 waves; block tile M=256, N=64.
// Each wave: 32(M) x 64(N) = 2x4 WMMA tiles -> per K-step 12 b128 loads feed
// 8 v_wmma_f32_16x16x32_bf16. All 4 B fragments live in distinct registers so
// the scheduler can clause the loads and overlap them with earlier WMMAs
// (partial s_wait_loadcnt instead of full drains).
// Operands are L2-resident (X 8MB, W 4MB per layer/dir); fragments are loaded
// straight from global with A and B both in contiguous k order (k = 16*half + s).
// ---------------------------------------------------------------------------
__global__ void k_gemm_xg(const unsigned short* __restrict__ Xb,
                          const unsigned short* __restrict__ Wb,
                          const float* __restrict__ bi,
                          const float* __restrict__ bh,
                          float* __restrict__ XG) {
    const int wave = threadIdx.x >> 5;
    const int lane = threadIdx.x & 31;
    const int half = lane >> 4;       // 0: lanes 0-15, 1: lanes 16-31
    const int m16  = lane & 15;
    const int M0 = blockIdx.y * 256 + wave * 32;
    const int N0 = blockIdx.x * 64;

    v8f acc[2][4];
#pragma unroll
    for (int mt = 0; mt < 2; ++mt)
#pragma unroll
        for (int nt = 0; nt < 4; ++nt)
#pragma unroll
            for (int r = 0; r < 8; ++r) acc[mt][nt][r] = 0.0f;

    const unsigned short* arow0 = Xb + (size_t)(M0 + m16) * E_;
    const unsigned short* arow1 = arow0 + (size_t)16 * E_;
    const unsigned short* brow  = Wb + (size_t)(N0 + m16) * E_;   // +nt*16 rows below

    for (int kk = 0; kk < E_; kk += 32) {
        // A fragment 16x32: half 0 -> K {kk..kk+15},
        //                   half 1 -> K {kk+16..kk+31}
        FragU a0, a1;
        {
            const float4* ap = reinterpret_cast<const float4*>(arow0 + kk + half * 16);
            a0.q[0] = ap[0];       // 8 bf16 @ kk + half*16
            a0.q[1] = ap[1];       // 8 bf16 @ kk + half*16 + 8
        }
        {
            const float4* ap = reinterpret_cast<const float4*>(arow1 + kk + half * 16);
            a1.q[0] = ap[0];
            a1.q[1] = ap[1];
        }
        // All 4 B fragments up front, in distinct registers.
        // B fragment 32x16: lane column n = W row (N0+nt*16+m16),
        // half selects K 0..15 vs 16..31 -> 32 contiguous bytes.
        FragU b[4];
#pragma unroll
        for (int nt = 0; nt < 4; ++nt) {
            const float4* bp = reinterpret_cast<const float4*>(
                brow + (size_t)(nt * 16) * E_ + kk + half * 16);
            b[nt].q[0] = bp[0];
            b[nt].q[1] = bp[1];
        }
#pragma unroll
        for (int nt = 0; nt < 4; ++nt) {
            acc[0][nt] = __builtin_amdgcn_wmma_f32_16x16x32_bf16(
                false, a0.v, false, b[nt].v, (short)0, acc[0][nt], false, false);
            acc[1][nt] = __builtin_amdgcn_wmma_f32_16x16x32_bf16(
                false, a1.v, false, b[nt].v, (short)0, acc[1][nt], false, false);
        }
    }

#pragma unroll
    for (int mt = 0; mt < 2; ++mt) {
#pragma unroll
        for (int nt = 0; nt < 4; ++nt) {
#pragma unroll
            for (int r = 0; r < 8; ++r) {
                int m = M0 + mt * 16 + half * 8 + r;  // C/D: vgpr r <-> M=r+8*half
                int n = N0 + nt * 16 + m16;           // N = lane&15
                XG[(size_t)m * G_ + n] = acc[mt][nt][r] + bi[n] + bh[n];
            }
        }
    }
}

// ---------------------------------------------------------------------------
// Persistent bidirectional LSTM recurrence for one layer.
// grid = 2 blocks (dir 0 = fwd, dir 1 = bwd), 1024 threads.
// h lives in LDS; the next step's 8KB gate-input row xg[t+1] is prefetched into
// a double-buffered LDS staging area with global_load_async_to_lds_b128
// (ASYNCcnt) while the 512-deep recurrent dot product runs.
// Each step every thread produces 2 of the 2048 gate pre-activations
// (g = xg[t] + h @ Whh^T) via coalesced reads of WhhT[k][j], then threads
// 0..511 apply the gates and update (h, c).
// ---------------------------------------------------------------------------
__global__ void k_lstm(const float* __restrict__ xg,     // [2][S][G] this layer
                       const float* __restrict__ whh_t,  // [2][H][G] this layer
                       const float* __restrict__ h0l,    // [2][H]
                       const float* __restrict__ c0l,    // [2][H]
                       float* __restrict__ Y) {          // [S][2H]
    __shared__ float hsh[H_];
    __shared__ float gsh[G_];
    __shared__ float xsh[2][G_];      // double-buffered async xg staging

    const int dir = blockIdx.x;
    const int tid = threadIdx.x;
    const float* xgd = xg    + (size_t)dir * S_ * G_;
    const float* wt  = whh_t + (size_t)dir * H_ * G_;

    if (tid < H_) hsh[tid] = h0l[dir * H_ + tid];
    float c = (tid < H_) ? c0l[dir * H_ + tid] : 0.0f;

    // Prime buffer 0 with step 0's gate row (lanes 0..511 copy 16B each).
    if (tid < H_) {
        const int s0 = dir ? (S_ - 1) : 0;
        async_ld_b128((unsigned)(unsigned long long)(uintptr_t)&xsh[0][tid * 4],
                      (unsigned long long)(uintptr_t)(xgd + (size_t)s0 * G_ + tid * 4));
    }
    __syncthreads();

    const int j1 = tid;          // gates i (0..511) / f (512..1023)
    const int j2 = tid + 1024;   // gates g (1024..1535) / o (1536..2047)

    for (int t = 0; t < S_; ++t) {
        const int s   = dir ? (S_ - 1 - t) : t;
        const int buf = t & 1;

        if (tid < H_) wait_asynccnt0();   // buffer `buf` fully landed in LDS
        __syncthreads();                  // ...and visible to all 32 waves

        // Kick off next step's xg row into the other buffer.
        if (t + 1 < S_ && tid < H_) {
            const int sn = dir ? (S_ - 2 - t) : (t + 1);
            async_ld_b128((unsigned)(unsigned long long)(uintptr_t)&xsh[buf ^ 1][tid * 4],
                          (unsigned long long)(uintptr_t)(xgd + (size_t)sn * G_ + tid * 4));
        }

        float a1 = xsh[buf][j1];
        float a2 = xsh[buf][j2];
#pragma unroll 8
        for (int k = 0; k < H_; ++k) {
            float hk = hsh[k];
            a1 = fmaf(hk, wt[(size_t)k * G_ + j1], a1);
            a2 = fmaf(hk, wt[(size_t)k * G_ + j2], a2);
            if ((k & 63) == 0 && k + 64 < H_) {
                __builtin_prefetch(wt + (size_t)(k + 64) * G_ + j1, 0, 3);
                __builtin_prefetch(wt + (size_t)(k + 64) * G_ + j2, 0, 3);
            }
        }
        gsh[j1] = a1;
        gsh[j2] = a2;
        __syncthreads();                   // dots done reading hsh; gsh visible
        if (tid < H_) {
            float iv = sigmoidf(gsh[tid]);
            float fv = sigmoidf(gsh[tid + 512]);
            float gv = tanhf(gsh[tid + 1024]);
            float ov = sigmoidf(gsh[tid + 1536]);
            c = fv * c + iv * gv;
            float h = ov * tanhf(c);
            hsh[tid] = h;
            Y[(size_t)s * (2 * H_) + dir * H_ + tid] = h;
        }
        __syncthreads();                   // new h visible before next step
    }
}

// ---------------------------------------------------------------------------
// Output projection: feats[s][t] = Y[s] . w_out[t] + b_out[t]
// ---------------------------------------------------------------------------
__global__ void k_proj(const float* __restrict__ Y, const float* __restrict__ w_out,
                       const float* __restrict__ b_out, float* __restrict__ feats) {
    int idx = blockIdx.x * blockDim.x + threadIdx.x;
    if (idx >= S_ * T_) return;
    int s = idx / T_, tt = idx % T_;
    const float* y = Y + (size_t)s * (2 * H_);
    const float* w = w_out + (size_t)tt * (2 * H_);
    float acc = b_out[tt];
#pragma unroll 8
    for (int k = 0; k < 2 * H_; ++k) acc = fmaf(y[k], w[k], acc);
    feats[idx] = acc;
}

// ---------------------------------------------------------------------------
// Viterbi (T=5): sequential scan + backtrack on one thread (trivially small).
// out[0] = path score; out[1..S] = best tag sequence (stored as float).
// ---------------------------------------------------------------------------
__global__ void k_viterbi(const float* __restrict__ feats, const float* __restrict__ trans,
                          int* __restrict__ bp, float* __restrict__ out) {
    if (threadIdx.x != 0 || blockIdx.x != 0) return;
    float fv[T_], nv[T_];
#pragma unroll
    for (int i = 0; i < T_; ++i) fv[i] = NEGINF;
    fv[3] = 0.0f;                                   // START tag
    for (int s = 0; s < S_; ++s) {
#pragma unroll
        for (int n = 0; n < T_; ++n) {
            float best = -3.0e38f; int bi = 0;
#pragma unroll
            for (int p = 0; p < T_; ++p) {
                float sc = fv[p] + trans[n * T_ + p];
                if (sc > best) { best = sc; bi = p; }
            }
            bp[s * T_ + n] = bi;
            nv[n] = best + feats[s * T_ + n];
        }
#pragma unroll
        for (int n = 0; n < T_; ++n) fv[n] = nv[n];
    }
    float best = -3.0e38f; int tag = 0;
#pragma unroll
    for (int p = 0; p < T_; ++p) {
        float sc = fv[p] + trans[4 * T_ + p];       // transitions[END]
        if (sc > best) { best = sc; tag = p; }
    }
    out[0] = best;
    for (int s = S_ - 1; s >= 1; --s) {
        out[1 + s] = (float)tag;
        tag = bp[s * T_ + tag];
    }
    out[1] = (float)tag;
}

// ---------------------------------------------------------------------------
extern "C" void kernel_launch(void* const* d_in, const int* in_sizes, int n_in,
                              void* d_out, int out_size, void* d_ws, size_t ws_size,
                              hipStream_t stream) {
    (void)in_sizes; (void)n_in; (void)out_size; (void)ws_size;

    const int*   sent  = (const int*)  d_in[0];
    const float* emb   = (const float*)d_in[1];
    const float* w_ih  = (const float*)d_in[2];   // [L,2,4H,E]
    const float* w_hh  = (const float*)d_in[3];   // [L,2,4H,H]
    const float* b_ih  = (const float*)d_in[4];   // [L,2,4H]
    const float* b_hh  = (const float*)d_in[5];   // [L,2,4H]
    const float* w_out = (const float*)d_in[6];   // [T,2H]
    const float* b_out = (const float*)d_in[7];   // [T]
    const float* trans = (const float*)d_in[8];   // [T,T]
    const float* h0    = (const float*)d_in[9];   // [L,2,H]
    const float* c0    = (const float*)d_in[10];  // [L,2,H]

    // Workspace layout
    char* ws = (char*)d_ws;
    size_t off = 0;
    unsigned short* Xbf  = (unsigned short*)(ws + off); off += (size_t)S_ * E_ * 2;           // 8 MB
    unsigned short* Wihb = (unsigned short*)(ws + off); off += (size_t)L_ * 2 * G_ * E_ * 2;  // 25 MB
    float*          WhhT = (float*)(ws + off);          off += (size_t)L_ * 2 * H_ * G_ * 4;  // 25 MB
    float*          XG   = (float*)(ws + off);          off += (size_t)2 * S_ * G_ * 4;       // 64 MB
    float*          Y    = (float*)(ws + off);          off += (size_t)S_ * 2 * H_ * 4;       // 16 MB
    float*          feats= (float*)(ws + off);          off += (size_t)S_ * T_ * 4;
    int*            bp   = (int*)(ws + off);            off += (size_t)S_ * T_ * 4;

    // One-time per call: embed + convert weights
    k_embed<<<(S_ * E_ + 255) / 256, 256, 0, stream>>>(sent, emb, Xbf);
    const int nih = L_ * 2 * G_ * E_;
    k_cvt_bf16<<<(nih + 255) / 256, 256, 0, stream>>>(w_ih, Wihb, nih);
    const int nhh = L_ * 2 * G_ * H_;
    k_whh_t<<<(nhh + 255) / 256, 256, 0, stream>>>(w_hh, WhhT);

    for (int l = 0; l < L_; ++l) {
        for (int d = 0; d < 2; ++d) {
            const unsigned short* Wb = Wihb + (size_t)(l * 2 + d) * G_ * E_;
            const float* bi = b_ih + (size_t)(l * 2 + d) * G_;
            const float* bh = b_hh + (size_t)(l * 2 + d) * G_;
            dim3 grid(G_ / 64, S_ / 256);   // (32, 16)
            k_gemm_xg<<<grid, 256, 0, stream>>>(Xbf, Wb, bi, bh, XG + (size_t)d * S_ * G_);
        }
        k_lstm<<<2, 1024, 0, stream>>>(XG,
                                       WhhT + (size_t)l * 2 * H_ * G_,
                                       h0 + (size_t)l * 2 * H_,
                                       c0 + (size_t)l * 2 * H_,
                                       Y);
        if (l < L_ - 1) {
            k_cvt_bf16<<<(S_ * E_ + 255) / 256, 256, 0, stream>>>(Y, Xbf, S_ * E_);
        }
    }

    k_proj<<<(S_ * T_ + 255) / 256, 256, 0, stream>>>(Y, w_out, b_out, feats);
    k_viterbi<<<1, 32, 0, stream>>>(feats, trans, bp, (float*)d_out);
}
